// SNN4_21878563406063
// MI455X (gfx1250) — compile-verified, matrix operand order checked by reading, not run on hardware
//
#include <hip/hip_runtime.h>

// ---------------------------------------------------------------------------
// MI455X (gfx1250) SNN forward.
// Roofline: GEMM1 = 50.8 GFLOP, x read = 508 MB (22us floor @23.3TB/s) ->
// f16 WMMA (f32 accum) so compute ~ memory floor. W1 panel (266KB f16) is
// DMA'd once per block into the 320KB WGP LDS by the Tensor Data Mover
// (tensor_load_to_lds + s_wait_tensorcnt). Scan: all recurrent history
// (9-deep rings for 16 batch rows) + W2/W3 live in LDS; per-step GEMMs via
// v_wmma_f32_16x16x32_f16.
// ---------------------------------------------------------------------------

typedef __attribute__((ext_vector_type(16))) _Float16 v16h;
typedef __attribute__((ext_vector_type(8)))  _Float16 v8h;
typedef __attribute__((ext_vector_type(8)))  float    v8f;
typedef __attribute__((ext_vector_type(4)))  unsigned int u32x4;
typedef __attribute__((ext_vector_type(8)))  int i32x8;
typedef __attribute__((ext_vector_type(4)))  int i32x4;

#define B_SZ   4096
#define T_SZ   50
#define D_IN   620
#define N1     200
#define N2     100
#define N3     11
#define MROWS  (B_SZ * T_SZ)      // 204800
#define K1P    640                // 620 padded to 32
#define N1P    208                // 200 padded to 16
#define K2P    224                // 200 padded to 32
#define N2P    112                // 100 padded to 16
#define K3P    128                // 100 padded to 32
#define N3P    16                 // 11 padded to 16

__device__ inline float sigmoid10(float v) { return 1.0f / (1.0f + __expf(-10.0f * v)); }
__device__ inline float swishf(float v)    { return v * sigmoid10(v); }
__device__ inline float actf(float v)      { return sigmoid10(v - 0.5f); }

// Load a 16x32 f16 fragment (A-layout; also valid for B == rows of W, since
// B[k][n] = W[n][k] means lane c loads W row n=c, K-contiguous).
// Lanes 0-15 (row=lane): K 0-7 & 16-23; lanes 16-31 (row=lane-16): K 8-15 & 24-31.
__device__ inline v16h frag_ld(const _Float16* base, int stride) {
    int lane = threadIdx.x & 31;
    int row  = lane & 15;
    int ko   = (lane >> 4) * 8;
    const _Float16* p = base + row * stride + ko;
    v8h lo = *(const v8h*)(p);
    v8h hi = *(const v8h*)(p + 16);
    v16h f;
#pragma unroll
    for (int i = 0; i < 8; ++i) { f[i] = lo[i]; f[i + 8] = hi[i]; }
    return f;
}

__device__ inline v8f wmma16(v16h a, v16h b, v8f c) {
    return __builtin_amdgcn_wmma_f32_16x16x32_f16(false, a, false, b, (short)0, c,
                                                  false, false);
}

// ---------------------------------------------------------------------------
// Pad + convert weight matrix [sr][sc] f32 -> [dr][dc] f16, zero padding.
// ---------------------------------------------------------------------------
__global__ void pad_f16_kernel(const float* __restrict__ src, _Float16* __restrict__ dst,
                               int sr, int sc, int dr, int dc) {
    int gid = blockIdx.x * 256 + threadIdx.x;
    if (gid >= dr * dc) return;
    int r = gid / dc, c = gid % dc;
    float v = (r < sr && c < sc) ? src[r * sc + c] : 0.0f;
    dst[gid] = (_Float16)v;
}

// ---------------------------------------------------------------------------
// GEMM1: H1[m][n] = swish(sum_k x[m][k] * W1[n][k]), m in [0,204800), n<200.
// Block tile: 128 M x 208 N, K chunks of 32. 8 waves; wave w owns M-subtile w,
// all 13 N tiles. The full W1 panel [208][640] f16 is TDM-loaded to LDS once.
// Dynamic LDS: As 8192 B + Bs 266240 B = 274432 B.
// ---------------------------------------------------------------------------
__global__ __launch_bounds__(256) void gemm1_kernel(const float* __restrict__ x,
                                                    const _Float16* __restrict__ W1p,
                                                    float* __restrict__ H1) {
    extern __shared__ char smem1[];
    _Float16* As = (_Float16*)smem1;          // [128][32]
    _Float16* Bs = As + 128 * 32;             // [208][640] full W1 panel

    const int tid  = threadIdx.x;
    const int wid  = tid >> 5;
    const int lane = tid & 31;
    const int m0   = blockIdx.x * 128;

    // --- Tensor Data Mover: DMA the whole W1 panel (266KB) into LDS ---
    if (wid == 0) {
        unsigned ldsoff = (unsigned)(uintptr_t)(void*)Bs;      // wave-relative LDS byte addr
        unsigned long long ga = (unsigned long long)(uintptr_t)W1p;
        // D# group0: count=1 | lds_addr | global_addr[56:0] | type=2 (bits 127:126)
        u32x4 g0 = { 1u, ldsoff,
                     (unsigned)(ga & 0xffffffffu),
                     (unsigned)((ga >> 32) & 0x01ffffffu) | (2u << 30) };
        // D# group1: data_size=2B (bits17:16); tensor_dim0=640 (79:48);
        // tensor_dim1=208 (111:80); tile_dim0=640 (127:112); tile_dim1=208 (143:128);
        // tensor_dim0_stride=640 (207:160); tensor_dim1_stride=0.
        i32x8 g1 = { (int)(1u << 16),          // data_size=1 -> 2 bytes
                     (int)((unsigned)K1P << 16),
                     (int)((unsigned)N1P << 16),
                     (int)((unsigned)K1P << 16),
                     (int)N1P,
                     (int)K1P,
                     0, 0 };
        i32x4 gz4 = { 0, 0, 0, 0 };
        i32x8 gz8 = { 0, 0, 0, 0, 0, 0, 0, 0 };
        __builtin_amdgcn_tensor_load_to_lds(g0, g1, gz4, gz4, gz8, 0);
        __builtin_amdgcn_s_wait_tensorcnt(0);
    }
    __syncthreads();

    v8f zero = {};
    v8f acc[13];
#pragma unroll
    for (int j = 0; j < 13; ++j) acc[j] = zero;

    for (int kc = 0; kc < K1P; kc += 32) {
        // Stage A: 128x32 f32 -> f16 (1024 float4 chunks, 4 per thread)
#pragma unroll
        for (int i = 0; i < 4; ++i) {
            int e  = i * 256 + tid;
            int r  = e >> 3;
            int c4 = (e & 7) * 4;
            int gk = kc + c4;
            float4 v = make_float4(0.f, 0.f, 0.f, 0.f);
            if (gk < D_IN) v = *(const float4*)(x + (size_t)(m0 + r) * D_IN + gk);
            _Float16* d = As + r * 32 + c4;
            d[0] = (_Float16)v.x; d[1] = (_Float16)v.y;
            d[2] = (_Float16)v.z; d[3] = (_Float16)v.w;
        }
        // Prefetch next K chunk of x (global_prefetch_b8)
        if (kc + 32 < D_IN)
            __builtin_prefetch(x + (size_t)(m0 + (tid >> 1)) * D_IN + kc + 32 + (tid & 1) * 16, 0, 1);
        __syncthreads();

        v16h af = frag_ld(As + (wid * 16) * 32, 32);
#pragma unroll
        for (int j = 0; j < 13; ++j) {
            v16h bf = frag_ld(Bs + (size_t)(j * 16) * K1P + kc, K1P);
            acc[j] = wmma16(af, bf, acc[j]);
        }
        __syncthreads();
    }

    // Epilogue: D layout -> row = wid*16 + (lane>=16)*8 + v, col = 16j + (lane&15)
    int mrow  = m0 + wid * 16 + (lane >> 4) * 8;
    int ncol0 = lane & 15;
#pragma unroll
    for (int j = 0; j < 13; ++j) {
        int n = j * 16 + ncol0;
        if (n < N1) {
#pragma unroll
            for (int v = 0; v < 8; ++v)
                H1[(size_t)(mrow + v) * N1 + n] = swishf(acc[j][v]);
        }
    }
}

// ---------------------------------------------------------------------------
// syn2[b][t][n] = sum_{k=0..9} K[k] * h1[b][t-k][n]   (rolling window)
// ---------------------------------------------------------------------------
__global__ __launch_bounds__(256) void syn2_kernel(const float* __restrict__ H1,
                                                   float* __restrict__ SYN2) {
    int gid = blockIdx.x * 256 + threadIdx.x;
    if (gid >= B_SZ * N1) return;
    int b = gid / N1, n = gid % N1;

    float kvv[10];
#pragma unroll
    for (int k = 0; k < 10; ++k) {
        float tt = (float)(k - 5);
        kvv[k] = (tt < 0.f) ? -0.5f * __expf(tt) : __expf(-0.5f * tt);
    }

    float win[10];
#pragma unroll
    for (int k = 0; k < 10; ++k) win[k] = 0.f;

    const float* src = H1 + (size_t)b * T_SZ * N1 + n;
    float* dst = SYN2 + (size_t)b * T_SZ * N1 + n;
    for (int t = 0; t < T_SZ; ++t) {
#pragma unroll
        for (int k = 9; k > 0; --k) win[k] = win[k - 1];
        win[0] = src[(size_t)t * N1];
        float s = 0.f;
#pragma unroll
        for (int k = 0; k < 10; ++k) s += kvv[k] * win[k];
        dst[(size_t)t * N1] = s;
    }
}

// ---------------------------------------------------------------------------
// Recurrent scan. Block = 16 batch rows, 256 threads (8 waves), all state in
// LDS (316 800 B < 320 KB WGP LDS). 9-deep rings indexed slot = t % 9;
// zero-init makes t-k<0 reads correct.
// ---------------------------------------------------------------------------
__global__ __launch_bounds__(256) void scan_kernel(const float* __restrict__ SYN2,
                                                   const _Float16* __restrict__ W2p,
                                                   const _Float16* __restrict__ W3p,
                                                   float* __restrict__ out) {
    extern __shared__ char smem[];
    float* SL2H = (float*)smem;               // [9][16][200]
    float* H2H  = SL2H + 9 * 16 * N1;         // [9][16][100]
    float* SL3H = H2H  + 9 * 16 * N2;         // [9][16][100]
    float* H3H  = SL3H + 9 * 16 * N2;         // [9][16][11]
    float* SL4H = H3H  + 9 * 16 * N3;         // [9][16][11]
    float* H2B  = SL4H + 9 * 16 * N3;         // [16][112]
    float* H3B  = H2B  + 16 * N2P;            // [16][16]
    _Float16* W2L = (_Float16*)(H3B + 16 * 16); // [112][224]
    _Float16* W3L = W2L + N2P * K2P;            // [16][128]
    _Float16* S2B = W3L + N3P * K3P;            // [16][224]
    _Float16* S3B = S2B + 16 * K2P;             // [16][128]

    const int tid  = threadIdx.x;
    const int wid  = tid >> 5;
    const int lane = tid & 31;
    const int b0   = blockIdx.x * 16;

    float kvv[10], ksv[10];
#pragma unroll
    for (int k = 0; k < 10; ++k) {
        float tt = (float)(k - 5);
        kvv[k] = (tt < 0.f) ? -0.5f * __expf(tt) : __expf(-0.5f * tt);
        ksv[k] = __expf(-0.5f * (float)k);
    }

    // Init: zero all history rings (contiguous), zero padded cols of S2B/S3B,
    // load W2/W3 into LDS.
    const int histN = 9 * 16 * (N1 + N2 + N2 + N3 + N3);   // 60768 floats
    for (int e = tid; e < histN; e += 256) SL2H[e] = 0.f;
    for (int e = tid; e < 16 * K2P; e += 256) S2B[e] = (_Float16)0.f;
    for (int e = tid; e < 16 * K3P; e += 256) S3B[e] = (_Float16)0.f;
    for (int e = tid; e < N2P * K2P; e += 256) W2L[e] = W2p[e];
    for (int e = tid; e < N3P * K3P; e += 256) W3L[e] = W3p[e];
    __syncthreads();

    float acc_out = 0.f;
    const int om = tid / N3, on = tid % N3;   // layer-4 mapping (tid < 176)

    for (int t = 0; t < T_SZ; ++t) {
        const int s = t % 9;

        // ---- layer 2 elementwise: mem2 = syn2_t - sum ks[k]*sl2(t-k) ----
        for (int e = tid; e < 16 * N1; e += 256) {
            int m = e / N1, n = e % N1;
            float syn = SYN2[((size_t)(b0 + m) * T_SZ + t) * N1 + n];
            float hs = 0.f;
#pragma unroll
            for (int k = 1; k < 10; ++k) {
                int sl = (t - k + 9) % 9;
                hs += ksv[k] * SL2H[(sl * 16 + m) * N1 + n];
            }
            float mem2 = syn - hs;
            float s2 = actf(mem2);
            SL2H[(s * 16 + m) * N1 + n] = s2 * mem2;
            S2B[m * K2P + n] = (_Float16)s2;
        }
        __syncthreads();

        // ---- GEMM2: h2 = swish(s2 @ W2^T), waves 0..6 own N-tiles ----
        if (wid < 7) {
            v8f acc = {};
#pragma unroll
            for (int kc = 0; kc < 7; ++kc) {
                v16h a = frag_ld(S2B + kc * 32, K2P);
                v16h b = frag_ld(W2L + (wid * 16) * K2P + kc * 32, K2P);
                acc = wmma16(a, b, acc);
            }
            int n = wid * 16 + (lane & 15);
            int mb = (lane >> 4) * 8;
            if (n < N2) {
#pragma unroll
                for (int v = 0; v < 8; ++v) H2B[(mb + v) * N2P + n] = swishf(acc[v]);
            }
        }
        __syncthreads();

        // ---- layer 3 elementwise ----
        for (int e = tid; e < 16 * N2; e += 256) {
            int m = e / N2, n = e % N2;
            float h2 = H2B[m * N2P + n];
            float mem3 = kvv[0] * h2;
#pragma unroll
            for (int k = 1; k < 10; ++k) {
                int sl = (t - k + 9) % 9;
                mem3 += kvv[k] * H2H[(sl * 16 + m) * N2 + n]
                      - ksv[k] * SL3H[(sl * 16 + m) * N2 + n];
            }
            float s3 = actf(mem3);
            H2H[(s * 16 + m) * N2 + n]  = h2;
            SL3H[(s * 16 + m) * N2 + n] = s3 * mem3;
            S3B[m * K3P + n] = (_Float16)s3;
        }
        __syncthreads();

        // ---- GEMM3: h3 = swish(s3 @ W3^T), wave 0 only ----
        if (wid == 0) {
            v8f acc = {};
#pragma unroll
            for (int kc = 0; kc < 4; ++kc) {
                v16h a = frag_ld(S3B + kc * 32, K3P);
                v16h b = frag_ld(W3L + kc * 32, K3P);
                acc = wmma16(a, b, acc);
            }
            int n = lane & 15;
            int mb = (lane >> 4) * 8;
            if (n < N3) {
#pragma unroll
                for (int v = 0; v < 8; ++v) H3B[(mb + v) * 16 + n] = swishf(acc[v]);
            }
        }
        __syncthreads();

        // ---- layer 4 elementwise + spike accumulation ----
        if (tid < 16 * N3) {
            float h3 = H3B[om * 16 + on];
            float mem4 = kvv[0] * h3;
#pragma unroll
            for (int k = 1; k < 10; ++k) {
                int sl = (t - k + 9) % 9;
                mem4 += kvv[k] * H3H[(sl * 16 + om) * N3 + on]
                      - ksv[k] * SL4H[(sl * 16 + om) * N3 + on];
            }
            float s4 = actf(mem4);
            H3H[(s * 16 + om) * N3 + on]  = h3;
            SL4H[(s * 16 + om) * N3 + on] = s4 * mem4;
            acc_out += s4;
        }
        __syncthreads();
    }

    if (tid < 16 * N3) out[(size_t)(b0 + om) * N3 + on] = acc_out * 0.02f;
}

// ---------------------------------------------------------------------------
extern "C" void kernel_launch(void* const* d_in, const int* in_sizes, int n_in,
                              void* d_out, int out_size, void* d_ws, size_t ws_size,
                              hipStream_t stream) {
    (void)in_sizes; (void)n_in; (void)out_size; (void)ws_size;
    const float* x  = (const float*)d_in[0];
    const float* W1 = (const float*)d_in[1];
    const float* W2 = (const float*)d_in[2];
    const float* W3 = (const float*)d_in[3];
    float* out = (float*)d_out;

    char* ws = (char*)d_ws;
    _Float16* W1p = (_Float16*)ws;                     // 208*640
    _Float16* W2p = W1p + N1P * K1P;                   // 112*224
    _Float16* W3p = W2p + N2P * K2P;                   // 16*128
    size_t wbytes = (size_t)(N1P * K1P + N2P * K2P + N3P * K3P) * sizeof(_Float16);
    wbytes = (wbytes + 255) & ~(size_t)255;
    float* H1   = (float*)(ws + wbytes);               // 204800*200 f32
    float* SYN2 = H1 + (size_t)MROWS * N1;             // 204800*200 f32

    pad_f16_kernel<<<(N1P * K1P + 255) / 256, 256, 0, stream>>>(W1, W1p, N1, D_IN, N1P, K1P);
    pad_f16_kernel<<<(N2P * K2P + 255) / 256, 256, 0, stream>>>(W2, W2p, N2, N1, N2P, K2P);
    pad_f16_kernel<<<(N3P * K3P + 255) / 256, 256, 0, stream>>>(W3, W3p, N3, N2, N3P, K3P);

    const int g1_lds = (128 * 32 + N1P * K1P) * (int)sizeof(_Float16);  // 274432
    (void)hipFuncSetAttribute((const void*)gemm1_kernel,
                              hipFuncAttributeMaxDynamicSharedMemorySize, g1_lds);
    gemm1_kernel<<<MROWS / 128, 256, g1_lds, stream>>>(x, W1p, H1);

    syn2_kernel<<<(B_SZ * N1 + 255) / 256, 256, 0, stream>>>(H1, SYN2);

    const int scan_lds = 316800;
    (void)hipFuncSetAttribute((const void*)scan_kernel,
                              hipFuncAttributeMaxDynamicSharedMemorySize, scan_lds);
    scan_kernel<<<B_SZ / 16, 256, scan_lds, stream>>>(SYN2, W2p, W3p, out);
}
